// LimitRegressor_5669356830899
// MI455X (gfx1250) — compile-verified, matrix operand order checked by reading, not run on hardware
//
#include <hip/hip_runtime.h>

// ---------------------------------------------------------------------------
// CDNA5 / gfx1250 fused DEQ-style MLP + fixed-point kernel.
//   F_x = W3^T tanh(W2^T tanh(W1^T x + b1) + b2) + b3
//   y   = fixed_point(F_x)  (converges to F_x within eps; iterated per-tile)
// GEMMs via V_WMMA_F32_16X16X32_BF16 (fp32 accumulate), activations in LDS.
// ---------------------------------------------------------------------------

typedef __attribute__((ext_vector_type(16))) __bf16 v16bf;
typedef __attribute__((ext_vector_type(4)))  __bf16 v4bf;
typedef __attribute__((ext_vector_type(8)))  float  v8f;

#define B_TOTAL 131072
#define D_IN    256
#define HID     512
#define D_OUT   128
#define BM      64          // batch rows per workgroup
#define NWAVES  8           // 256 threads
#define ALPHA_C 0.2f
#define EPS_ELEM 8.8388348e-8f   // 1e-6 / sqrt(128): per-element bound => row-norm < eps
#define K_MAX_C 1000

__device__ __forceinline__ float fast_tanh(float x) {
#if __has_builtin(__builtin_amdgcn_tanhf)
  return __builtin_amdgcn_tanhf(x);
#elif __has_builtin(__builtin_amdgcn_tanh_f32)
  return __builtin_amdgcn_tanh_f32(x);
#else
  float r;
  asm("v_tanh_f32 %0, %1" : "=v"(r) : "v"(x));
  return r;
#endif
}

// Load a 16x32 bf16 fragment (A layout; also B layout with N playing M's role).
// base = &mat[row0][k0], row-major, 'stride' elements per row.
// Per ISA 7.12.2: lanes 0-15 hold rows 0-15 with K in {kb..kb+7, kb+16..kb+23},
// kb = 8*(lane>=16). Two contiguous 16-byte loads per lane.
__device__ __forceinline__ v16bf load_frag(const __bf16* base, int stride, int lane) {
  const int m  = lane & 15;
  const int kb = (lane >> 4) << 3;
  const __bf16* p = base + m * stride + kb;
  v16bf r;
  ((uint4*)&r)[0] = *(const uint4*)(p);        // K = kb .. kb+7
  ((uint4*)&r)[1] = *(const uint4*)(p + 16);   // K = kb+16 .. kb+23
  return r;
}

// Epilogue for hidden layers: h = tanh(acc + bias), stored bf16 into LDS.
// C/D layout: lane holds N = n0+(lane&15); VGPR r holds M = m0 + 8*(lane>>4) + r.
__device__ __forceinline__ void store_h_tile(__bf16* ldsH, int stride, int m0, int n0,
                                             const v8f& acc, const float* bias, int lane) {
  const int n     = n0 + (lane & 15);
  const int mbase = m0 + ((lane >> 4) << 3);
  const float bv = bias[n];
#pragma unroll
  for (int r = 0; r < 8; ++r) {
    float v = fast_tanh(acc[r] + bv);
    ldsH[(mbase + r) * stride + n] = (__bf16)v;
  }
}

// One dense layer: (BM x K) bf16 in LDS  @  Wt (N x K, bf16, global/L2) -> tanh -> LDS.
__device__ __forceinline__ void mlp_layer(const __bf16* ldsA, int K,
                                          const __bf16* __restrict__ Wt,
                                          const float* __restrict__ bias, int N,
                                          __bf16* ldsH, int wave, int lane) {
  const int nt_per_wave = (N / 16) / NWAVES;   // 4 for N=512
  for (int j = 0; j < nt_per_wave; ++j) {
    const int n0 = (wave * nt_per_wave + j) * 16;
    v8f acc[4];
#pragma unroll
    for (int mt = 0; mt < 4; ++mt)
#pragma unroll
      for (int r = 0; r < 8; ++r) acc[mt][r] = 0.0f;

    for (int k0 = 0; k0 < K; k0 += 32) {
      v16bf bf = load_frag(Wt + n0 * K + k0, K, lane);
#pragma unroll
      for (int mt = 0; mt < 4; ++mt) {
        v16bf af = load_frag(ldsA + (mt * 16) * K + k0, K, lane);
        acc[mt] = __builtin_amdgcn_wmma_f32_16x16x32_bf16(
            false, af, false, bf, (short)0, acc[mt], false, false);
      }
    }
#pragma unroll
    for (int mt = 0; mt < 4; ++mt)
      store_h_tile(ldsH, N, mt * 16, n0, acc[mt], bias, lane);
  }
}

__global__ __launch_bounds__(256)
void deq_mlp_fused_kernel(const float* __restrict__ x,
                          const __bf16* __restrict__ W1t, const float* __restrict__ b1,
                          const __bf16* __restrict__ W2t, const float* __restrict__ b2,
                          const __bf16* __restrict__ W3t, const float* __restrict__ b3,
                          float* __restrict__ out) {
  __shared__ __align__(16) __bf16 lds_h1[BM * HID];   // 64 KB
  __shared__ __align__(16) __bf16 lds_h2[BM * HID];   // 64 KB (aliased as x-stage)
  __bf16* lds_x = lds_h2;                             // x tile dead before layer2 writes h2

  const int tid  = threadIdx.x;
  const int wave = tid >> 5;
  const int lane = tid & 31;
  const int row0 = blockIdx.x * BM;

  // ---- Stage x tile (BM x D_IN) fp32 -> bf16 in LDS -------------------------
  {
    const float4* xg4 = (const float4*)(x + (size_t)row0 * D_IN);
    v4bf* lx4 = (v4bf*)lds_x;
    for (int i = tid; i < (BM * D_IN) / 4; i += 256) {
      float4 f = xg4[i];
      v4bf h;
      h[0] = (__bf16)f.x; h[1] = (__bf16)f.y; h[2] = (__bf16)f.z; h[3] = (__bf16)f.w;
      lx4[i] = h;
    }
  }
  __syncthreads();

  // ---- Layer 1: h1 = tanh(x @ W1 + b1) --------------------------------------
  mlp_layer(lds_x, D_IN, W1t, b1, HID, lds_h1, wave, lane);
  __syncthreads();

  // ---- Layer 2: h2 = tanh(h1 @ W2 + b2) -------------------------------------
  mlp_layer(lds_h1, HID, W2t, b2, HID, lds_h2, wave, lane);
  __syncthreads();

  // ---- Layer 3: F = h2 @ W3 + b3, then fixed point --------------------------
  {
    const int n0 = wave * 16;                 // D_OUT=128 -> 8 n-tiles, 1 per wave
    v8f acc[4];
#pragma unroll
    for (int mt = 0; mt < 4; ++mt)
#pragma unroll
      for (int r = 0; r < 8; ++r) acc[mt][r] = 0.0f;

    for (int k0 = 0; k0 < HID; k0 += 32) {
      v16bf bf = load_frag(W3t + n0 * HID + k0, HID, lane);
#pragma unroll
      for (int mt = 0; mt < 4; ++mt) {
        v16bf af = load_frag(lds_h2 + (mt * 16) * HID + k0, HID, lane);
        acc[mt] = __builtin_amdgcn_wmma_f32_16x16x32_bf16(
            false, af, false, bf, (short)0, acc[mt], false, false);
      }
    }

    const int n  = n0 + (lane & 15);
    const int hi = (lane >> 4) << 3;
    const float bv = b3[n];

    float F[4][8], y[4][8];
#pragma unroll
    for (int mt = 0; mt < 4; ++mt)
#pragma unroll
      for (int r = 0; r < 8; ++r) { F[mt][r] = acc[mt][r] + bv; y[mt][r] = 0.0f; }

    // y_{k+1} = y_k + alpha*tanh(F - y_k); per-element bound |step| < eps/sqrt(D)
    // guarantees row-norm step < eps (tile-local early exit; <= reference iters).
    for (int it = 0; it < K_MAX_C; ++it) {
      float mx = 0.0f;
#pragma unroll
      for (int mt = 0; mt < 4; ++mt)
#pragma unroll
        for (int r = 0; r < 8; ++r) {
          float s = ALPHA_C * fast_tanh(F[mt][r] - y[mt][r]);
          y[mt][r] += s;
          mx = fmaxf(mx, fabsf(s));
        }
      if (mx < EPS_ELEM) break;
    }

#pragma unroll
    for (int mt = 0; mt < 4; ++mt)
#pragma unroll
      for (int r = 0; r < 8; ++r) {
        int row = row0 + mt * 16 + hi + r;
        out[(size_t)row * D_OUT + n] = y[mt][r];
      }
  }
}

// Convert fp32 weight [K][N] -> bf16 transposed [N][K] (so B fragments are
// two contiguous b128 loads per lane). Tiny (<1 MB total), runs once per call.
__global__ void cvt_transpose_bf16(const float* __restrict__ src,
                                   __bf16* __restrict__ dst, int K, int N) {
  int idx = blockIdx.x * blockDim.x + threadIdx.x;
  if (idx >= K * N) return;
  int k = idx / N;
  int n = idx - k * N;
  dst[n * K + k] = (__bf16)src[idx];
}

extern "C" void kernel_launch(void* const* d_in, const int* in_sizes, int n_in,
                              void* d_out, int out_size, void* d_ws, size_t ws_size,
                              hipStream_t stream) {
  (void)in_sizes; (void)n_in; (void)out_size; (void)ws_size;
  const float* x  = (const float*)d_in[0];
  const float* W1 = (const float*)d_in[1];
  const float* b1 = (const float*)d_in[2];
  const float* W2 = (const float*)d_in[3];
  const float* b2 = (const float*)d_in[4];
  const float* W3 = (const float*)d_in[5];
  const float* b3 = (const float*)d_in[6];
  float* out = (float*)d_out;

  __bf16* W1t = (__bf16*)d_ws;                 // 512 x 256 -> 256 KB
  __bf16* W2t = W1t + (size_t)D_IN * HID;      // 512 x 512 -> 512 KB
  __bf16* W3t = W2t + (size_t)HID * HID;       // 128 x 512 -> 128 KB

  cvt_transpose_bf16<<<(D_IN * HID + 255) / 256, 256, 0, stream>>>(W1, W1t, D_IN, HID);
  cvt_transpose_bf16<<<(HID * HID + 255) / 256, 256, 0, stream>>>(W2, W2t, HID, HID);
  cvt_transpose_bf16<<<(HID * D_OUT + 255) / 256, 256, 0, stream>>>(W3, W3t, HID, D_OUT);

  deq_mlp_fused_kernel<<<B_TOTAL / BM, 256, 0, stream>>>(x, W1t, b1, W2t, b2, W3t, b3, out);
}
